// CustomRNN_31181462569479
// MI455X (gfx1250) — matrix operand order repaired and checked by reading, not compiled, not run on hardware
//
#include <hip/hip_runtime.h>
#include <hip/hip_bf16.h>
#include <math.h>

// ---------------------------------------------------------------------------
// GRU (2048 x [256,512]) + FC + log-softmax for MI455X (gfx1250, wave32)
// bf16 WMMA everywhere; A panels moved into LDS by the Tensor Data Mover.
// ---------------------------------------------------------------------------

typedef __attribute__((ext_vector_type(16))) __bf16 v16bf;
typedef __attribute__((ext_vector_type(8)))  __bf16 v8bf;
typedef __attribute__((ext_vector_type(8)))  float  v8f;
typedef __attribute__((ext_vector_type(4)))  float  v4f;
typedef __attribute__((ext_vector_type(4)))  unsigned int v4u;
typedef __attribute__((ext_vector_type(8)))  int    v8i;
typedef __attribute__((ext_vector_type(4)))  int    v4i;

#define SEQ   2048
#define BATCH 256
#define DIN   512
#define DH    512
#define DOUT  1000
#define NPAD  1024

// GEMM tiling: block = 256 threads = 8 waves, macro tile 64(M) x 128(N),
// each wave computes a 16x64 strip (4 WMMA C tiles), K stepped by 32.
#define BM 64
#define BN 128

// TDM A-panel: per 256-k chunk, 64 rows x 256 bf16 (512B) + 16B pad/row
#define CHUNK_K       256
#define KT_PER_CHUNK  8
#define LDS_ROW_BYTES 528                 // 512B data + 16B pad (bank stride 4)
#define LDS_PANEL_BF16 (BM * LDS_ROW_BYTES / 2)

// fp32-staged fallback path (fc kernel): 32 bf16 + 8 pad per row
#define LDS_PITCH 40

#if defined(__gfx1250__) && __has_builtin(__builtin_amdgcn_tensor_load_to_lds) && \
    __has_builtin(__builtin_amdgcn_s_wait_tensorcnt)
#define USE_TDM 1
#else
#define USE_TDM 0
#endif

// ---------------------------------------------------------------------------
// Weight packing: B operand pre-swizzled so each lane's 16-bf16 WMMA fragment
// is contiguous: lane = n%16 + 16*((k%32 / 8) & 1), elem = k%8 + 8*(k%32 >= 16),
// pos = ((nt*KT + kt)*32 + lane)*16 + elem
// ---------------------------------------------------------------------------
__device__ __forceinline__ size_t swzB(int k, int n, int KT) {
  int kt = k >> 5, ko = k & 31;
  int nt = n >> 4, no = n & 15;
  int lane = no + (((ko >> 3) & 1) ? 16 : 0);
  int e    = (ko & 7) + ((ko >> 4) << 3);
  return (((size_t)(nt * KT + kt) * 32 + lane) << 4) + e;
}

__global__ __launch_bounds__(256) void pack_zr_kernel(
    const float* __restrict__ Wz, const float* __restrict__ Uz,
    const float* __restrict__ Wr, const float* __restrict__ Ur,
    __bf16* __restrict__ Bzr) {
  int idx = blockIdx.x * 256 + threadIdx.x;      // N(1024) x K(1024)
  if (idx >= 1024 * 1024) return;
  int n = idx >> 10, k = idx & 1023;
  float v;
  if (n < DH) v = (k < DIN) ? Wz[k * DH + n] : Uz[(k - DIN) * DH + n];
  else { int nn = n - DH; v = (k < DIN) ? Wr[k * DH + nn] : Ur[(k - DIN) * DH + nn]; }
  Bzr[swzB(k, n, 32)] = (__bf16)v;
}

__global__ __launch_bounds__(256) void pack_h_kernel(
    const float* __restrict__ Wh, const float* __restrict__ Uh,
    __bf16* __restrict__ Bh) {
  int idx = blockIdx.x * 256 + threadIdx.x;      // N(512) x K(1024)
  if (idx >= 512 * 1024) return;
  int n = idx >> 10, k = idx & 1023;
  float v = (k < DIN) ? Wh[k * DH + n] : Uh[(k - DIN) * DH + n];
  Bh[swzB(k, n, 32)] = (__bf16)v;
}

__global__ __launch_bounds__(256) void pack_fc_kernel(
    const float* __restrict__ Wfc, __bf16* __restrict__ Bfc) {
  int idx = blockIdx.x * 256 + threadIdx.x;      // N(1024) x K(512)
  if (idx >= 1024 * 512) return;
  int n = idx >> 9, k = idx & 511;
  float v = (n < DOUT) ? Wfc[k * DOUT + n] : 0.0f;
  Bfc[swzB(k, n, 16)] = (__bf16)v;
}

// one-time fp32 -> bf16 conversion of the whole input sequence
__global__ __launch_bounds__(256) void convert_x_kernel(
    const float* __restrict__ x, __bf16* __restrict__ Xb, long long n8) {
  long long idx = (long long)blockIdx.x * 256 + threadIdx.x;
  if (idx >= n8) return;
  const float* s = x + idx * 8;
  v4f a = *(const v4f*)s;
  v4f b = *(const v4f*)(s + 4);
  v8bf o;
#pragma unroll
  for (int i = 0; i < 4; ++i) { o[i] = (__bf16)a[i]; o[4 + i] = (__bf16)b[i]; }
  *(v8bf*)(Xb + idx * 8) = o;
}

__global__ __launch_bounds__(256) void setup_misc_kernel(
    const float* __restrict__ bz,  const float* __restrict__ buz,
    const float* __restrict__ br,  const float* __restrict__ bur,
    const float* __restrict__ bh,  const float* __restrict__ buh,
    const float* __restrict__ bfc,
    float* __restrict__ bias_zr, float* __restrict__ bias_h,
    float* __restrict__ bias_fc, float* __restrict__ Hf,
    __bf16* __restrict__ Hb) {
  int idx = blockIdx.x * 256 + threadIdx.x;      // 512 blocks -> 131072 threads
  if (idx < BATCH * DH) { Hf[idx] = 0.0f; Hb[idx] = (__bf16)0.0f; }
  if (idx < 1024) {
    bias_zr[idx] = (idx < DH) ? (bz[idx] + buz[idx]) : (br[idx - DH] + bur[idx - DH]);
    bias_fc[idx] = (idx < DOUT) ? bfc[idx] : 0.0f;
  }
  if (idx < DH) bias_h[idx] = bh[idx] + buh[idx];
}

// ---------------------------------------------------------------------------
// TDM descriptor: load 2D tile [64 rows x 256 bf16], row stride 512 elements,
// into LDS at lds_off with 16B padding appended per 512B row.
// Bitfields per CDNA5 ISA.
// This toolchain exposes the 6-arg builtin:
//   (uint32x4 g0, int32x8 g1, int32x4 g2, int32x4 g3, int32x8 extra, i32 cpol)
// ---------------------------------------------------------------------------
#if USE_TDM
__device__ __forceinline__ void tdm_load_panel(const __bf16* src, unsigned lds_off) {
  unsigned long long ga = (unsigned long long)(uintptr_t)src;
  v4u g0 = { 1u,                                          // count=1, no gather
             lds_off,                                     // lds_addr (bytes)
             (unsigned)(ga & 0xffffffffu),                // global_addr[31:0]
             (unsigned)((ga >> 32) & 0x01ffffffu) | (2u << 30) }; // addr[56:32] | type=2
  v8i g1 = {
    (int)((1u << 16) | (1u << 20) | (6u << 22) | (3u << 25)),
    //      data=2B     pad_en      interval=128DW  amount=4DW
    (int)(256u << 16),                 // tensor_dim0[15:0]=256 (bits 63:48)
    (int)(64u << 16),                  // tensor_dim0[31:16]=0, tensor_dim1[15:0]=64
    (int)(256u << 16),                 // tensor_dim1[31:16]=0, tile_dim0=256
    (int)(64u),                        // tile_dim1=64, tile_dim2=0
    (int)(512u),                       // tensor_dim0_stride[31:0]=512
    0, 0                               // stride hi / dim1_stride = 0
  };
  v4i gz4 = {0, 0, 0, 0};
  v8i gz8 = {0, 0, 0, 0, 0, 0, 0, 0};
  __builtin_amdgcn_tensor_load_to_lds(g0, g1, gz4, gz4, gz8, 0);
}
#endif

// ---------------------------------------------------------------------------
// TDM-fed WMMA GEMM core (bf16 A sources, row pitch 512).
// A = [A0 (k 0..511) | A1 (k 512..1023)], K=1024, each wave fills 4 C tiles.
// ---------------------------------------------------------------------------
__device__ __forceinline__ void gemm_tdm(
    const __bf16* __restrict__ A0, const __bf16* __restrict__ A1,
    const __bf16* __restrict__ Bsw, int KT,
    int nt_base_wave, __bf16* lds, v8f c[4]) {
  const int tid  = threadIdx.x;
  const int lane = tid & 31;
  const int wave = tid >> 5;
  const int wm   = wave & 3;
  const int half = lane >> 4;
  const int row  = wm * 16 + (lane & 15);
  const char* lbase = (const char*)lds + row * LDS_ROW_BYTES + half * 16;

  for (int chunk = 0; chunk < 4; ++chunk) {
    const __bf16* src = (chunk < 2) ? (A0 + (size_t)chunk * CHUNK_K)
                                    : (A1 + (size_t)(chunk - 2) * CHUNK_K);
    __syncthreads();                       // previous chunk fully consumed
#if USE_TDM
    if (wave == 0) {
      tdm_load_panel(src, (unsigned)(uintptr_t)lds);
      __builtin_amdgcn_s_wait_tensorcnt((short)0);   // s_wait_tensorcnt 0
    }
#else
    {   // cooperative bf16 panel copy: 64 rows x 512B, 128B per thread
      int r = tid >> 2, seg = (tid & 3) * 128;
      const char* gsrc = (const char*)(src + (size_t)r * 512) + seg;
      char* ldst = (char*)lds + r * LDS_ROW_BYTES + seg;
#pragma unroll
      for (int j = 0; j < 8; ++j)
        *(v8bf*)(ldst + j * 16) = *(const v8bf*)(gsrc + j * 16);
    }
#endif
    __syncthreads();                       // panel visible to all waves

#pragma unroll
    for (int kl = 0; kl < KT_PER_CHUNK; ++kl) {
      const int kt = chunk * KT_PER_CHUNK + kl;
      // A fragment per ISA 7.12.2: lane<16 -> K{0..7,16..23}, lane>=16 -> +8
      const char* ab = lbase + kl * 64;
      v8bf alo = *(const v8bf*)ab;
      v8bf ahi = *(const v8bf*)(ab + 32);
      v16bf afrag = __builtin_shufflevector(alo, ahi,
          0, 1, 2, 3, 4, 5, 6, 7, 8, 9, 10, 11, 12, 13, 14, 15);
#pragma unroll
      for (int i = 0; i < 4; ++i) {
        const __bf16* bp =
            Bsw + ((((size_t)(nt_base_wave + i) * KT + kt) << 5) + lane) * 16;
        v8bf blo = *(const v8bf*)bp;
        v8bf bhi = *(const v8bf*)(bp + 8);
        v16bf bfrag = __builtin_shufflevector(blo, bhi,
            0, 1, 2, 3, 4, 5, 6, 7, 8, 9, 10, 11, 12, 13, 14, 15);
        c[i] = __builtin_amdgcn_wmma_f32_16x16x32_bf16(
            false, afrag, false, bfrag, (short)0, c[i], false, false);
      }
      if (kt + 1 < KT) {
        const __bf16* np =
            Bsw + ((((size_t)nt_base_wave * KT + (kt + 1)) << 5) + lane) * 16;
        __builtin_prefetch(np, 0, 1);      // global_prefetch next K slice of B
      }
    }
  }
}

// ---------------------------------------------------------------------------
// fp32-staged WMMA core (fc only): A fp32 pitch 512, optional ReLU in stage.
// ---------------------------------------------------------------------------
__device__ __forceinline__ void gemm_fp32(
    const float* __restrict__ A0, int K, const __bf16* __restrict__ Bsw,
    bool relu, int gm_base, int nt_base_wave, __bf16* lds, v8f c[4]) {
  const int tid  = threadIdx.x;
  const int lane = tid & 31;
  const int wave = tid >> 5;
  const int wm   = wave & 3;
  const int KT   = K >> 5;
  const int half = lane >> 4;
  const int lds_base = (wm * 16 + (lane & 15)) * LDS_PITCH + half * 8;

  for (int kt = 0; kt < KT; ++kt) {
    __syncthreads();
    {
      int r = tid >> 2, kc = (tid & 3) << 3;
      const float* src = A0 + (size_t)(gm_base + r) * 512 + (kt << 5) + kc;
      v4f a = *(const v4f*)src;
      v4f b = *(const v4f*)(src + 4);
      v8bf o;
#pragma unroll
      for (int i = 0; i < 4; ++i) {
        float va = relu ? fmaxf(a[i], 0.0f) : a[i];
        float vb = relu ? fmaxf(b[i], 0.0f) : b[i];
        o[i] = (__bf16)va; o[4 + i] = (__bf16)vb;
      }
      *(v8bf*)(lds + r * LDS_PITCH + kc) = o;
    }
    __syncthreads();
    v8bf alo = *(const v8bf*)(lds + lds_base);
    v8bf ahi = *(const v8bf*)(lds + lds_base + 16);
    v16bf afrag = __builtin_shufflevector(alo, ahi,
        0, 1, 2, 3, 4, 5, 6, 7, 8, 9, 10, 11, 12, 13, 14, 15);
#pragma unroll
    for (int i = 0; i < 4; ++i) {
      const __bf16* bp =
          Bsw + ((((size_t)(nt_base_wave + i) * KT + kt) << 5) + lane) * 16;
      v8bf blo = *(const v8bf*)bp;
      v8bf bhi = *(const v8bf*)(bp + 8);
      v16bf bfrag = __builtin_shufflevector(blo, bhi,
          0, 1, 2, 3, 4, 5, 6, 7, 8, 9, 10, 11, 12, 13, 14, 15);
      c[i] = __builtin_amdgcn_wmma_f32_16x16x32_bf16(
          false, afrag, false, bfrag, (short)0, c[i], false, false);
    }
  }
}

// ---------------------------------------------------------------------------
// Per-timestep kernels
// ---------------------------------------------------------------------------
__global__ __launch_bounds__(256) void gru_zr_kernel(
    const __bf16* __restrict__ Xb, const __bf16* __restrict__ Hb,
    const float* __restrict__ Hf, const __bf16* __restrict__ Bzr,
    const float* __restrict__ bias_zr, float* __restrict__ Z,
    __bf16* __restrict__ RHb, int t) {
  __shared__ __attribute__((aligned(16))) __bf16 lds[LDS_PANEL_BF16];
  const int lane = threadIdx.x & 31, wave = threadIdx.x >> 5;
  const int wm = wave & 3, wn = wave >> 2;
  const int gm_base = blockIdx.y * BM;
  const int nt_base = blockIdx.x * 8 + wn * 4;
  const __bf16* A0 = Xb + ((size_t)t * BATCH + gm_base) * DIN;
  const __bf16* A1 = Hb + (size_t)gm_base * DH;

  v8f c[4];
#pragma unroll
  for (int i = 0; i < 4; ++i) c[i] = (v8f){0.f,0.f,0.f,0.f,0.f,0.f,0.f,0.f};

  gemm_tdm(A0, A1, Bzr, 32, nt_base, lds, c);

  const int half = lane >> 4, nl = lane & 15;
#pragma unroll
  for (int i = 0; i < 4; ++i) {
    int n = (nt_base + i) * 16 + nl;
    float bval = bias_zr[n];
#pragma unroll
    for (int v = 0; v < 8; ++v) {
      int m = gm_base + wm * 16 + v + 8 * half;
      float s = 1.0f / (1.0f + __expf(-(c[i][v] + bval)));
      if (n < DH) {
        Z[(size_t)m * DH + n] = s;                        // z gate
      } else {
        int nn = n - DH;                                  // r gate -> r*h (bf16)
        RHb[(size_t)m * DH + nn] = (__bf16)(s * Hf[(size_t)m * DH + nn]);
      }
    }
  }
}

__global__ __launch_bounds__(256) void gru_h_kernel(
    const __bf16* __restrict__ Xb, const __bf16* __restrict__ RHb,
    const __bf16* __restrict__ Bh, const float* __restrict__ bias_h,
    const float* __restrict__ Z, float* __restrict__ Hf,
    __bf16* __restrict__ Hb, int t) {
  __shared__ __attribute__((aligned(16))) __bf16 lds[LDS_PANEL_BF16];
  const int lane = threadIdx.x & 31, wave = threadIdx.x >> 5;
  const int wm = wave & 3, wn = wave >> 2;
  const int gm_base = blockIdx.y * BM;
  const int nt_base = blockIdx.x * 8 + wn * 4;
  const __bf16* A0 = Xb + ((size_t)t * BATCH + gm_base) * DIN;
  const __bf16* A1 = RHb + (size_t)gm_base * DH;

  v8f c[4];
#pragma unroll
  for (int i = 0; i < 4; ++i) c[i] = (v8f){0.f,0.f,0.f,0.f,0.f,0.f,0.f,0.f};

  gemm_tdm(A0, A1, Bh, 32, nt_base, lds, c);

  const int half = lane >> 4, nl = lane & 15;
#pragma unroll
  for (int i = 0; i < 4; ++i) {
    int n = (nt_base + i) * 16 + nl;
    float bval = bias_h[n];
#pragma unroll
    for (int v = 0; v < 8; ++v) {
      int m = gm_base + wm * 16 + v + 8 * half;
      size_t idx = (size_t)m * DH + n;
      float ht = tanhf(c[i][v] + bval);
      float ho = Hf[idx];
      float z  = Z[idx];
      float hn = ho + z * (ht - ho);       // (1-z)*h + z*h~
      Hf[idx] = hn;
      Hb[idx] = (__bf16)hn;
    }
  }
}

// relu(H) @ Wfc + bfc  -> logits[256, 1024 (padded)]
__global__ __launch_bounds__(256) void fc_kernel(
    const float* __restrict__ Hf, const __bf16* __restrict__ Bfc,
    const float* __restrict__ bias_fc, float* __restrict__ logits) {
  __shared__ __attribute__((aligned(16))) __bf16 lds[BM * LDS_PITCH];
  const int lane = threadIdx.x & 31, wave = threadIdx.x >> 5;
  const int wm = wave & 3, wn = wave >> 2;
  const int gm_base = blockIdx.y * BM;
  const int nt_base = blockIdx.x * 8 + wn * 4;

  v8f c[4];
#pragma unroll
  for (int i = 0; i < 4; ++i) c[i] = (v8f){0.f,0.f,0.f,0.f,0.f,0.f,0.f,0.f};

  gemm_fp32(Hf, DH, Bfc, true /*relu*/, gm_base, nt_base, lds, c);

  const int half = lane >> 4, nl = lane & 15;
#pragma unroll
  for (int i = 0; i < 4; ++i) {
    int n = (nt_base + i) * 16 + nl;
    float bval = bias_fc[n];
#pragma unroll
    for (int v = 0; v < 8; ++v) {
      int m = gm_base + wm * 16 + v + 8 * half;
      logits[(size_t)m * NPAD + n] = c[i][v] + bval;
    }
  }
}

// log_softmax over axis=1 (the BATCH axis) of [1, 256, 1000]
__global__ __launch_bounds__(256) void logsoftmax_kernel(
    const float* __restrict__ logits, float* __restrict__ out) {
  int j = blockIdx.x;        // class 0..999
  int b = threadIdx.x;       // batch 0..255
  __shared__ float red[256];
  float v = logits[(size_t)b * NPAD + j];
  red[b] = v;
  __syncthreads();
  for (int s = 128; s > 0; s >>= 1) {
    if (b < s) red[b] = fmaxf(red[b], red[b + s]);
    __syncthreads();
  }
  float mx = red[0];
  __syncthreads();
  red[b] = __expf(v - mx);
  __syncthreads();
  for (int s = 128; s > 0; s >>= 1) {
    if (b < s) red[b] += red[b + s];
    __syncthreads();
  }
  float lse = mx + __logf(red[0]);
  out[(size_t)b * DOUT + j] = v - lse;
}

// ---------------------------------------------------------------------------
extern "C" void kernel_launch(void* const* d_in, const int* in_sizes, int n_in,
                              void* d_out, int out_size, void* d_ws, size_t ws_size,
                              hipStream_t stream) {
  (void)in_sizes; (void)n_in; (void)out_size; (void)ws_size;
  const float* x   = (const float*)d_in[0];
  const float* Wz  = (const float*)d_in[1];  const float* bz  = (const float*)d_in[2];
  const float* Uz  = (const float*)d_in[3];  const float* buz = (const float*)d_in[4];
  const float* Wr  = (const float*)d_in[5];  const float* br  = (const float*)d_in[6];
  const float* Ur  = (const float*)d_in[7];  const float* bur = (const float*)d_in[8];
  const float* Wh  = (const float*)d_in[9];  const float* bh  = (const float*)d_in[10];
  const float* Uh  = (const float*)d_in[11]; const float* buh = (const float*)d_in[12];
  const float* Wfc = (const float*)d_in[13]; const float* bfc = (const float*)d_in[14];
  float* out = (float*)d_out;

  char* p = (char*)d_ws;
  auto alloc = [&](size_t bytes) -> char* {
    char* r = p;
    p += (bytes + 255) & ~(size_t)255;
    return r;
  };
  __bf16* Xb      = (__bf16*)alloc((size_t)SEQ * BATCH * DIN * 2);   // 512 MB
  __bf16* Bzr     = (__bf16*)alloc((size_t)1024 * 1024 * 2);
  __bf16* Bh      = (__bf16*)alloc((size_t)1024 * 512 * 2);
  __bf16* Bfc     = (__bf16*)alloc((size_t)512 * 1024 * 2);
  float*  bias_zr = (float*)alloc(1024 * 4);
  float*  bias_h  = (float*)alloc(512 * 4);
  float*  bias_fc = (float*)alloc(1024 * 4);
  float*  Hf      = (float*)alloc((size_t)BATCH * DH * 4);
  __bf16* Hb      = (__bf16*)alloc((size_t)BATCH * DH * 2);
  __bf16* RHb     = (__bf16*)alloc((size_t)BATCH * DH * 2);
  float*  Z       = (float*)alloc((size_t)BATCH * DH * 4);
  float*  logits  = (float*)alloc((size_t)BATCH * NPAD * 4);

  // one-time packing / conversion / init
  long long n8 = (long long)SEQ * BATCH * DIN / 8;
  convert_x_kernel<<<(unsigned)(n8 / 256), 256, 0, stream>>>(x, Xb, n8);
  pack_zr_kernel<<<4096, 256, 0, stream>>>(Wz, Uz, Wr, Ur, Bzr);
  pack_h_kernel <<<2048, 256, 0, stream>>>(Wh, Uh, Bh);
  pack_fc_kernel<<<2048, 256, 0, stream>>>(Wfc, Bfc);
  setup_misc_kernel<<<512, 256, 0, stream>>>(bz, buz, br, bur, bh, buh, bfc,
                                             bias_zr, bias_h, bias_fc, Hf, Hb);

  // sequential recurrence: 2 dependent GEMM kernels per timestep
  for (int t = 0; t < SEQ; ++t) {
    gru_zr_kernel<<<dim3(8, 4), 256, 0, stream>>>(Xb, Hb, Hf, Bzr, bias_zr, Z, RHb, t);
    gru_h_kernel <<<dim3(4, 4), 256, 0, stream>>>(Xb, RHb, Bh, bias_h, Z, Hf, Hb, t);
  }

  fc_kernel<<<dim3(8, 4), 256, 0, stream>>>(Hf, Bfc, bias_fc, logits);
  logsoftmax_kernel<<<DOUT, 256, 0, stream>>>(logits, out);
}
